// MultHeadSelfAttentionDecoder_85890755985476
// MI455X (gfx1250) — matrix operand order repaired and checked by reading, not compiled, not run on hardware
//
#include <hip/hip_runtime.h>
#include <hip/hip_bf16.h>

typedef __attribute__((ext_vector_type(16))) __bf16 v16bf;
typedef __attribute__((ext_vector_type(8)))  __bf16 v8bf;
typedef __attribute__((ext_vector_type(8)))  float  v8f;
typedef __attribute__((ext_vector_type(4))) unsigned int u32x4;
typedef __attribute__((ext_vector_type(4))) int  i32x4;
typedef __attribute__((ext_vector_type(8))) int  i32x8;

#define B_  2
#define S_  2048
#define D_  1024
#define H_  16
#define HD_ 64
#define KC_ 256   // staged K-chunk (columns of the 64-row weight panel)

static __device__ __forceinline__ __bf16 f2bf(float f) { return (__bf16)f; }

// Build a 16x32 bf16 A-fragment (one contiguous 16-element run per lane, k = 16g + s).
static __device__ __forceinline__ v16bf load_a(const __bf16* row, int k0, int g) {
    union { v16bf v; v8bf h[2]; } u;
    u.h[0] = *(const v8bf*)(row + k0 + 16 * g);
    u.h[1] = *(const v8bf*)(row + k0 + 16 * g + 8);
    return u.v;
}

#if __has_builtin(__builtin_amdgcn_tensor_load_to_lds)
#define HAVE_TDM 1
#else
#define HAVE_TDM 0
#endif

// Issue (without waiting) a TDM load of a rows x cols bf16 tile into LDS.
// Called by ONE wave. On toolchains without the builtin, does a synchronous
// cooperative copy instead.
static __device__ __forceinline__ void tdm_issue(__bf16* lds_dst, const __bf16* gsrc,
                                                 int rows, int cols, int row_stride_elems) {
#if HAVE_TDM
    const unsigned            lds_addr = (unsigned)(size_t)lds_dst; // flat[31:0] = LDS offset
    const unsigned long long  ga       = (unsigned long long)(size_t)gsrc;
    u32x4 g0;
    g0[0] = 1u;                                                  // count=1, user D#
    g0[1] = lds_addr;                                            // lds_addr (bytes)
    g0[2] = (unsigned)ga;                                        // global_addr lo
    g0[3] = (unsigned)((ga >> 32) & 0x01FFFFFFu) | (2u << 30);   // global_addr hi | type=2
    i32x8 g1;
    g1[0] = (int)(1u << 16);                                     // data_size=1 (2B), wg_mask=0
    g1[1] = (int)(((unsigned)cols & 0xFFFFu) << 16);             // tensor_dim0 lo16 @bit48
    g1[2] = (int)((((unsigned)cols >> 16) & 0xFFFFu)
                | (((unsigned)rows & 0xFFFFu) << 16));           // tensor_dim0 hi / tensor_dim1 lo
    g1[3] = (int)((((unsigned)rows >> 16) & 0xFFFFu)
                | (((unsigned)cols & 0xFFFFu) << 16));           // tensor_dim1 hi / tile_dim0
    g1[4] = (int)((unsigned)rows & 0xFFFFu);                     // tile_dim1 (tile_dim2=0)
    g1[5] = (int)(unsigned)row_stride_elems;                     // tensor_dim0_stride lo32
    g1[6] = 0;                                                   // stride hi / dim1_stride
    g1[7] = 0;
    i32x4 z4 = {0, 0, 0, 0};
#if __clang_major__ >= 23
    i32x8 z8 = {0, 0, 0, 0, 0, 0, 0, 0};
    __builtin_amdgcn_tensor_load_to_lds(g0, g1, z4, z4, z8, 0);
#else
    __builtin_amdgcn_tensor_load_to_lds(g0, g1, z4, z4, 0);
#endif
#else
    const int lane = threadIdx.x & 31;
    const int vecs = rows * cols / 8;
    for (int i = lane; i < vecs; i += 32) {
        const int rr = (i * 8) / cols;
        const int cc = (i * 8) % cols;
        *(v8bf*)(lds_dst + rr * cols + cc) =
            *(const v8bf*)(gsrc + (size_t)rr * row_stride_elems + cc);
    }
#endif
}

static __device__ __forceinline__ void tdm_wait_le1() {
#if HAVE_TDM
    __builtin_amdgcn_s_wait_tensorcnt(1);
#endif
}
static __device__ __forceinline__ void tdm_wait_0() {
#if HAVE_TDM
    __builtin_amdgcn_s_wait_tensorcnt(0);
#endif
}

// ---------------------------------------------------------------------------
// Kernel 0: per-batch valid length
// ---------------------------------------------------------------------------
__global__ void lengths_kernel(const int* __restrict__ mask, int* __restrict__ lengths) {
    __shared__ int red[256];
    const int b = blockIdx.x, t = threadIdx.x;
    int s = 0;
    for (int i = t; i < S_; i += 256) s += mask[b * S_ + i];
    red[t] = s; __syncthreads();
    for (int off = 128; off > 0; off >>= 1) {
        if (t < off) red[t] += red[t + off];
        __syncthreads();
    }
    if (t == 0) lengths[b] = red[0];
}

// ---------------------------------------------------------------------------
// Kernel 1a: elementwise f32 -> bf16
// ---------------------------------------------------------------------------
__global__ void cvt_bf16_kernel(const float* __restrict__ in, __bf16* __restrict__ out, int n) {
    int i = blockIdx.x * 256 + threadIdx.x;
    if (i < n) out[i] = f2bf(in[i]);
}

// Kernel 1b: Wq/Wk/Wv [H][D][HD] f32 -> [H][HD][D] bf16 (transpose per head)
__global__ void wqkv_t_kernel(const float* __restrict__ w, __bf16* __restrict__ out) {
    int i = blockIdx.x * 256 + threadIdx.x;       // H_*D_*HD_ elements
    int e = i % HD_;
    int d = (i / HD_) % D_;
    int h = i / (HD_ * D_);
    out[((size_t)h * HD_ + e) * D_ + d] = f2bf(w[i]);
}

// ---------------------------------------------------------------------------
// Kernel 2: QKV projection + fused RoPE.
// grid = (BS/128, 3*H), block = 128 (4 waves). Each wave owns 32 rows; the
// 64x256 weight panels are TDM-staged into double-buffered LDS: the DMA for
// panel i+1 flies while all 4 waves compute on panel i.
// q,k -> [b][h][s][e]; v -> [b][h][e][s].
// ---------------------------------------------------------------------------
__global__ void __launch_bounds__(128) qkv_rope_kernel(
        const __bf16* __restrict__ xb,
        const __bf16* __restrict__ Wqt,
        const __bf16* __restrict__ Wkt,
        const __bf16* __restrict__ Wvt,
        __bf16* __restrict__ qb,
        __bf16* __restrict__ kb,
        __bf16* __restrict__ vT) {
    __shared__ __bf16 wpanel[2][HD_ * KC_];    // 2 x 32 KB, ping-pong

    const int tid  = threadIdx.x;
    const int wid  = tid >> 5;
    const int lane = tid & 31;
    const int g    = lane >> 4;
    const int ml   = lane & 15;
    const int mt   = blockIdx.x;            // 128-row tile of flattened (b,s)
    const int gy   = blockIdx.y;            // 0..47
    const int sel  = gy / H_;               // 0=q 1=k 2=v
    const int h    = gy % H_;

    const __bf16* wt = ((sel == 0) ? Wqt : (sel == 1) ? Wkt : Wvt) + (size_t)h * HD_ * D_;

    const int row0 = mt * 128 + wid * 32;
    const __bf16* ar0 = xb + (size_t)(row0 + ml) * D_;
    const __bf16* ar1 = xb + (size_t)(row0 + 16 + ml) * D_;

    v8f c[2][4] = {{v8f{}, v8f{}, v8f{}, v8f{}}, {v8f{}, v8f{}, v8f{}, v8f{}}};

    if (tid < 32) tdm_issue(wpanel[0], wt, HD_, KC_, D_);

    for (int kbase = 0; kbase < D_; kbase += KC_) {
        const int cur = (kbase / KC_) & 1;
        __syncthreads();                    // readers of wpanel[cur^1] are done
        if (tid < 32) {
            if (kbase + KC_ < D_) {
                tdm_issue(wpanel[cur ^ 1], wt + kbase + KC_, HD_, KC_, D_);
                tdm_wait_le1();             // older TDM (wpanel[cur]) complete
            } else {
                tdm_wait_0();
            }
        }
        __syncthreads();                    // publish wpanel[cur]
        const __bf16* wp = wpanel[cur];
        for (int k0 = 0; k0 < KC_; k0 += 32) {
            if (kbase + k0 + 32 < D_) {
                __builtin_prefetch(ar0 + kbase + k0 + 32, 0, 3);
                __builtin_prefetch(ar1 + kbase + k0 + 32, 0, 3);
            }
            const v16bf a0 = load_a(ar0, kbase + k0, g);
            const v16bf a1 = load_a(ar1, kbase + k0, g);
            #pragma unroll
            for (int j = 0; j < 4; j++) {
                // B[k][n] = Wt[e = j*16+n][d]: contiguous 16 bf16 from LDS
                const v16bf bfr = *(const v16bf*)(wp + (j * 16 + ml) * KC_ + k0 + 16 * g);
                c[0][j] = __builtin_amdgcn_wmma_f32_16x16x32_bf16(
                    false, a0, false, bfr, (short)0, c[0][j], false, false);
                c[1][j] = __builtin_amdgcn_wmma_f32_16x16x32_bf16(
                    false, a1, false, bfr, (short)0, c[1][j], false, false);
            }
        }
    }

    const int bidx  = row0 / S_;
    const int sloc0 = row0 - bidx * S_;

    if (sel == 2) {
        // v: transposed store, 8 consecutive s per lane -> one 16B store
        #pragma unroll
        for (int mm = 0; mm < 2; mm++) {
            const int sbase = sloc0 + 16 * mm + 8 * g;
            #pragma unroll
            for (int j = 0; j < 4; j++) {
                const int e = j * 16 + ml;
                v8bf pv;
                #pragma unroll
                for (int r = 0; r < 8; r++) pv[r] = f2bf(c[mm][j][r]);
                *(v8bf*)(vT + ((size_t)(bidx * H_ + h) * HD_ + e) * S_ + sbase) = pv;
            }
        }
    } else {
        __bf16* outb = ((sel == 0) ? qb : kb) + ((size_t)bidx * H_ + h) * (size_t)S_ * HD_;
        #pragma unroll
        for (int mm = 0; mm < 2; mm++) {
            #pragma unroll
            for (int j = 0; j < 4; j++) {
                const int   e     = j * 16 + ml;
                const int   em    = e & 31;
                const float theta = __expf(-(float)em * (9.210340371976184f / 32.0f));
                #pragma unroll
                for (int r = 0; r < 8; r++) {
                    const int   s   = sloc0 + 16 * mm + 8 * g + r;
                    const float ang = (float)s * theta;
                    const float rot = (j < 2) ? -c[mm][j + 2][r] : c[mm][j - 2][r];
                    const float val = c[mm][j][r] * __cosf(ang) + rot * __sinf(ang);
                    outb[(size_t)s * HD_ + e] = f2bf(val);
                }
            }
        }
    }
}

// ---------------------------------------------------------------------------
// Kernel 3: flash-style attention, one wave per (b,h, 16-row q tile).
// ---------------------------------------------------------------------------
__global__ void attn_kernel(const __bf16* __restrict__ qb,
                            const __bf16* __restrict__ kb,
                            const __bf16* __restrict__ vT,
                            const int* __restrict__ lengths,
                            __bf16* __restrict__ ob) {
    __shared__ __bf16 lds_p[16 * 32];

    const int lane = threadIdx.x & 31;
    const int g    = lane >> 4;
    const int ml   = lane & 15;
    const int mt   = blockIdx.x;            // S/16 tiles
    const int bh   = blockIdx.y;            // B*H
    const int b    = bh / H_;
    const int h    = bh % H_;
    const int m0   = mt * 16;
    const int len  = lengths[b];

    const __bf16* qh = qb + (size_t)bh * S_ * HD_;
    const __bf16* kh = kb + (size_t)bh * S_ * HD_;
    const __bf16* vh = vT + (size_t)bh * HD_ * S_;

    const __bf16* qrow = qh + (size_t)(m0 + ml) * HD_;
    const v16bf aq0 = load_a(qrow, 0, g);
    const v16bf aq1 = load_a(qrow, 32, g);

    v8f o[4] = {v8f{}, v8f{}, v8f{}, v8f{}};
    float mrow[8], lrow[8];
    #pragma unroll
    for (int r = 0; r < 8; r++) { mrow[r] = -1e30f; lrow[r] = 0.0f; }

    int hi = m0 + 16; if (hi > len) hi = len;
    const int   nch   = (hi + 31) / 32;
    const float scale = 0.125f;             // 1/sqrt(64)

    for (int kc = 0; kc < nch; kc++) {
        const int t0 = kc * 32;

        v8f sc[2];
        #pragma unroll
        for (int jt = 0; jt < 2; jt++) {
            const __bf16* kp = kh + (size_t)(t0 + jt * 16 + ml) * HD_;
            const v16bf b0 = *(const v16bf*)(kp + 16 * g);
            const v16bf b1 = *(const v16bf*)(kp + 32 + 16 * g);
            v8f s = {};
            s = __builtin_amdgcn_wmma_f32_16x16x32_bf16(false, aq0, false, b0, (short)0, s, false, false);
            s = __builtin_amdgcn_wmma_f32_16x16x32_bf16(false, aq1, false, b1, (short)0, s, false, false);
            sc[jt] = s;
        }

        // mask + scale + row max (half-wave reduce)
        float mnew[8];
        #pragma unroll
        for (int r = 0; r < 8; r++) {
            const int srow = m0 + r + 8 * g;
            float best = mrow[r];
            #pragma unroll
            for (int jt = 0; jt < 2; jt++) {
                const int  t  = t0 + jt * 16 + ml;
                const bool ok = (t <= srow) && (t < len);
                float v = ok ? sc[jt][r] * scale : -1e30f;
                sc[jt][r] = v;
                best = fmaxf(best, v);
            }
            best = fmaxf(best, __shfl_xor(best, 1, 16));
            best = fmaxf(best, __shfl_xor(best, 2, 16));
            best = fmaxf(best, __shfl_xor(best, 4, 16));
            best = fmaxf(best, __shfl_xor(best, 8, 16));
            mnew[r] = best;
        }

        // p = exp(s - mnew) (0 where masked), update l, rescale o, stash p
        #pragma unroll
        for (int r = 0; r < 8; r++) {
            const int srow = m0 + r + 8 * g;
            float psum = 0.0f;
            #pragma unroll
            for (int jt = 0; jt < 2; jt++) {
                const int  t  = t0 + jt * 16 + ml;
                const bool ok = (t <= srow) && (t < len);
                const float p = ok ? __expf(sc[jt][r] - mnew[r]) : 0.0f;
                psum += p;
                lds_p[(r + 8 * g) * 32 + jt * 16 + ml] = f2bf(p);
            }
            psum += __shfl_xor(psum, 1, 16);
            psum += __shfl_xor(psum, 2, 16);
            psum += __shfl_xor(psum, 4, 16);
            psum += __shfl_xor(psum, 8, 16);
            const float alpha = __expf(mrow[r] - mnew[r]);
            lrow[r] = lrow[r] * alpha + psum;
            mrow[r] = mnew[r];
            #pragma unroll
            for (int j = 0; j < 4; j++) o[j][r] *= alpha;
        }
        __syncthreads();

        // P as A-fragment from LDS, then PV WMMAs
        union { v16bf v; v8bf hh[2]; } up;
        up.hh[0] = *(const v8bf*)(lds_p + ml * 32 + 16 * g);
        up.hh[1] = *(const v8bf*)(lds_p + ml * 32 + 16 * g + 8);
        const v16bf ap = up.v;

        #pragma unroll
        for (int j = 0; j < 4; j++) {
            // B[k][n] = vT[e = j*16+n][t = t0+K]: contiguous 16 bf16 per lane
            const v16bf bv = *(const v16bf*)(vh + (size_t)(j * 16 + ml) * S_ + t0 + 16 * g);
            o[j] = __builtin_amdgcn_wmma_f32_16x16x32_bf16(
                false, ap, false, bv, (short)0, o[j], false, false);
        }
        __syncthreads();
    }

    // normalize, write bf16 attn output (B, S, H*hd)
    #pragma unroll
    for (int r = 0; r < 8; r++) {
        const float inv = (lrow[r] > 0.0f) ? 1.0f / lrow[r] : 0.0f;
        const int s = m0 + r + 8 * g;
        #pragma unroll
        for (int j = 0; j < 4; j++) {
            const int d = h * HD_ + j * 16 + ml;
            ob[((size_t)b * S_ + s) * D_ + d] = f2bf(o[j][r] * inv);
        }
    }
}

// ---------------------------------------------------------------------------
// Kernel 4: out = attn_out @ W^T.  4-wave blocks, double-buffered TDM panels.
// ---------------------------------------------------------------------------
__global__ void __launch_bounds__(128) proj_kernel(
        const __bf16* __restrict__ ob,
        const __bf16* __restrict__ Wb,
        float* __restrict__ out) {
    __shared__ __bf16 wpanel[2][HD_ * KC_];    // 2 x 32 KB, ping-pong

    const int tid  = threadIdx.x;
    const int wid  = tid >> 5;
    const int lane = tid & 31;
    const int g    = lane >> 4;
    const int ml   = lane & 15;
    const int mt   = blockIdx.x;            // 32 tiles of 128 rows
    const int nt   = blockIdx.y;            // 16 col groups of 64
    const int row0 = mt * 128 + wid * 32;

    const __bf16* ar0 = ob + (size_t)(row0 + ml) * D_;
    const __bf16* ar1 = ob + (size_t)(row0 + 16 + ml) * D_;
    const __bf16* wrows = Wb + (size_t)(nt * 64) * D_;   // rows n of W (cols of W^T)

    v8f c[2][4] = {{v8f{}, v8f{}, v8f{}, v8f{}}, {v8f{}, v8f{}, v8f{}, v8f{}}};

    if (tid < 32) tdm_issue(wpanel[0], wrows, HD_, KC_, D_);

    for (int kbase = 0; kbase < D_; kbase += KC_) {
        const int cur = (kbase / KC_) & 1;
        __syncthreads();
        if (tid < 32) {
            if (kbase + KC_ < D_) {
                tdm_issue(wpanel[cur ^ 1], wrows + kbase + KC_, HD_, KC_, D_);
                tdm_wait_le1();
            } else {
                tdm_wait_0();
            }
        }
        __syncthreads();
        const __bf16* wp = wpanel[cur];
        for (int k0 = 0; k0 < KC_; k0 += 32) {
            if (kbase + k0 + 32 < D_) {
                __builtin_prefetch(ar0 + kbase + k0 + 32, 0, 3);
                __builtin_prefetch(ar1 + kbase + k0 + 32, 0, 3);
            }
            const v16bf a0 = load_a(ar0, kbase + k0, g);
            const v16bf a1 = load_a(ar1, kbase + k0, g);
            #pragma unroll
            for (int j = 0; j < 4; j++) {
                const v16bf bfr = *(const v16bf*)(wp + (j * 16 + ml) * KC_ + k0 + 16 * g);
                c[0][j] = __builtin_amdgcn_wmma_f32_16x16x32_bf16(
                    false, a0, false, bfr, (short)0, c[0][j], false, false);
                c[1][j] = __builtin_amdgcn_wmma_f32_16x16x32_bf16(
                    false, a1, false, bfr, (short)0, c[1][j], false, false);
            }
        }
    }

    #pragma unroll
    for (int mm = 0; mm < 2; mm++) {
        #pragma unroll
        for (int r = 0; r < 8; r++) {
            const int row = row0 + 16 * mm + r + 8 * g;
            #pragma unroll
            for (int j = 0; j < 4; j++)
                out[(size_t)row * D_ + nt * 64 + j * 16 + ml] = c[mm][j][r];
        }
    }
}

// ---------------------------------------------------------------------------
extern "C" void kernel_launch(void* const* d_in, const int* in_sizes, int n_in,
                              void* d_out, int out_size, void* d_ws, size_t ws_size,
                              hipStream_t stream) {
    const float* x  = (const float*)d_in[0];
    const int*   pm = (const int*)  d_in[1];
    const float* Wq = (const float*)d_in[2];
    const float* Wk = (const float*)d_in[3];
    const float* Wv = (const float*)d_in[4];
    const float* W  = (const float*)d_in[5];

    const size_t NX = (size_t)B_ * S_ * D_;       // 4,194,304
    const size_t NW = (size_t)H_ * HD_ * D_;      // 1,048,576
    const size_t NQ = (size_t)B_ * H_ * S_ * HD_; // 4,194,304

    __bf16* xb  = (__bf16*)d_ws;
    __bf16* Wqt = xb  + NX;
    __bf16* Wkt = Wqt + NW;
    __bf16* Wvt = Wkt + NW;
    __bf16* Wb  = Wvt + NW;           // D*D, layout preserved
    __bf16* qb  = Wb  + (size_t)D_ * D_;
    __bf16* kb  = qb  + NQ;
    __bf16* vT  = kb  + NQ;           // [b][h][e][s]
    __bf16* ob  = vT  + NQ;           // [b][s][h*hd]
    int* lengths = (int*)(ob + NX);

    lengths_kernel<<<B_, 256, 0, stream>>>(pm, lengths);
    cvt_bf16_kernel<<<(int)(NX / 256), 256, 0, stream>>>(x, xb, (int)NX);
    cvt_bf16_kernel<<<(int)(((size_t)D_ * D_) / 256), 256, 0, stream>>>(W, Wb, D_ * D_);
    wqkv_t_kernel<<<(int)(NW / 256), 256, 0, stream>>>(Wq, Wqt);
    wqkv_t_kernel<<<(int)(NW / 256), 256, 0, stream>>>(Wk, Wkt);
    wqkv_t_kernel<<<(int)(NW / 256), 256, 0, stream>>>(Wv, Wvt);

    qkv_rope_kernel<<<dim3((B_ * S_) / 128, 3 * H_), 128, 0, stream>>>(
        xb, Wqt, Wkt, Wvt, qb, kb, vT);
    attn_kernel<<<dim3(S_ / 16, B_ * H_), 32, 0, stream>>>(
        qb, kb, vT, lengths, ob);
    proj_kernel<<<dim3((B_ * S_) / 128, D_ / 64), 128, 0, stream>>>(
        ob, Wb, (float*)d_out);
}
